// FoldableWindowAttn4D_Train_48146583388185
// MI455X (gfx1250) — matrix operand order repaired and checked by reading, not compiled, not run on hardware
//
#include <hip/hip_runtime.h>
#include <hip/hip_bf16.h>

// ---------------------------------------------------------------------------
// FoldableWindowAttn4D fully fused for gfx1250 (MI455X), wave32 + WMMA.
//   x: (4,128,256,256) f32 ; cpe_w (128,1,3,3) ; cpe_b (128)
//   Wq/Wk/Wv: (2,4,32,128) ; Wo: (2,4,128,32)
//   out: (4,128,256,256) f32
// One workgroup (512 thr = 16 waves) per 16x16 window; wave w owns window row w
// (pixels p = 16w..16w+15). All GEMMs via v_wmma_f32_16x16x32_f16.
// Per-branch QKV weights are staged into LDS with async load-to-LDS
// (overlapped with CPE / attention), waited with s_wait_asynccnt.
// ---------------------------------------------------------------------------

typedef __attribute__((ext_vector_type(16))) _Float16 v16h;
typedef __attribute__((ext_vector_type(8)))  _Float16 v8h;
typedef __attribute__((ext_vector_type(8)))  float    v8f;
typedef __attribute__((__vector_size__(16))) int      v4i;

#define KSTR 136                    // K_lds row stride (halves): [256 pix][136]
#define VSTR 264                    // V_lds row stride (halves): [128 ch ][264]
#define K_HALVES (256 * KSTR)       // 34816
#define V_HALVES (128 * VSTR)       // 33792
#define W_HALVES 49152              // per-branch QKV weights: 3*4*32*128
#define SMEM_BYTES ((K_HALVES + V_HALVES + W_HALVES) * 2)   // 235,520 B

#if defined(__gfx1250__) && \
    __has_builtin(__builtin_amdgcn_global_load_async_to_lds_b128)
#define HAVE_ASYNC_LDS 1
#define AS1 __attribute__((address_space(1)))
#define AS3 __attribute__((address_space(3)))
#else
#define HAVE_ASYNC_LDS 0
#endif

static __device__ __forceinline__ v8f wmma16(v16h a, v16h b, v8f c) {
  // D = A(16x32 f16) x B(32x16 f16) + C(16x16 f32)
  return __builtin_amdgcn_wmma_f32_16x16x32_f16(false, a, false, b, (short)0, c,
                                                false, false);
}
static __device__ __forceinline__ v16h cat16(const _Float16* lo,
                                             const _Float16* hi) {
  v8h a = *(const v8h*)lo;
  v8h b = *(const v8h*)hi;
  v16h r;
#pragma unroll
  for (int i = 0; i < 8; ++i) { r[i] = a[i]; r[i + 8] = b[i]; }
  return r;
}
static __device__ __forceinline__ v8h shflx16_h8(v8h v) {
  union { v8h h; int i[4]; } u;
  u.h = v;
#pragma unroll
  for (int k = 0; k < 4; ++k) u.i[k] = __shfl_xor(u.i[k], 16);
  return u.h;
}
static __device__ __forceinline__ int clampi(int v, int lo, int hi) {
  return v < lo ? lo : (v > hi ? hi : v);
}

// Issue the per-branch QKV weight copy global->LDS (6144 16-byte chunks,
// 12 per thread).  Async when available (ASYNCcnt), else plain copy.
static __device__ __forceinline__ void issue_weight_copy(
    const _Float16* __restrict__ wall, _Float16* ldsw, int br, int tid) {
#pragma unroll
  for (int i = 0; i < 12; ++i) {
    const int chunk = tid + i * 512;                 // < 6144
    const _Float16* src = wall + br * W_HALVES + chunk * 8;
    _Float16* dst = ldsw + chunk * 8;
#if HAVE_ASYNC_LDS
    __builtin_amdgcn_global_load_async_to_lds_b128(
        (AS1 v4i*)(unsigned long long)(size_t)src,
        (AS3 v4i*)(unsigned int)(size_t)dst, 0, 0);
#else
    *(v8h*)dst = *(const v8h*)src;
#endif
  }
}
static __device__ __forceinline__ void wait_weight_copy() {
#if HAVE_ASYNC_LDS
  asm volatile("s_wait_asynccnt 0x0" ::: "memory");
#endif
}

// ---------------------------------------------------------------------------
// Weight conversion: f32 -> f16 into workspace.
//   ws[0      .. 98303]  : WQKV [br][proj][h][o=32][d=128]
//   ws[98304  ..131071]  : WO   [br][h][d=128][o=32]  (same layout as input)
// ---------------------------------------------------------------------------
__global__ void fwa_cvt_weights(const float* __restrict__ wq,
                                const float* __restrict__ wk,
                                const float* __restrict__ wv,
                                const float* __restrict__ wo,
                                _Float16* __restrict__ out) {
  int t = blockIdx.x * blockDim.x + threadIdx.x;   // 131072 total
  if (t < 98304) {
    int d = t & 127;
    int r = t >> 7;
    int o = r & 31;  r >>= 5;
    int hh = r & 3;  r >>= 2;
    int proj = r % 3;
    int br = r / 3;
    const float* src = (proj == 0) ? wq : (proj == 1) ? wk : wv;
    out[t] = (_Float16)src[(((br * 4 + hh) * 32 + o) << 7) + d];
  } else {
    out[t] = (_Float16)wo[t - 98304];
  }
}

// ---------------------------------------------------------------------------
// Main fused kernel.
// ---------------------------------------------------------------------------
__global__ __launch_bounds__(512, 1)
void fwa_main(const float* __restrict__ x, const float* __restrict__ cpe_w,
              const float* __restrict__ cpe_b,
              const _Float16* __restrict__ wall,   // ws base (QKV + Wo, f16)
              float* __restrict__ out) {
  extern __shared__ _Float16 smem[];
  _Float16* K_lds = smem;                       // pixel-major  [256][136]
  _Float16* V_lds = smem + K_HALVES;            // channel-major[128][264]
  _Float16* W_lds = smem + K_HALVES + V_HALVES; // per-branch QKV weights

  const int tid = threadIdx.x;
  const int w = tid >> 5;          // wave id == window row (0..15)
  const int lane = tid & 31;
  const int n = lane & 15;         // column-in-tile == window col of owned pixel
  const int h = lane >> 4;         // lane half
  const int p = w * 16 + n;        // owned pixel within window

  const int wb = blockIdx.x;       // 1024 windows
  const int b = wb >> 8;
  const int wy = (wb >> 4) & 15;
  const int wx = wb & 15;
  const int gy = wy * 16 + w;      // global row of owned pixel
  const int gx = wx * 16 + n;      // global col of owned pixel

  // Stage branch-0 QKV weights; overlaps with the CPE phase below.
  issue_weight_copy(wall, W_lds, 0, tid);

  // ---------------- CPE: xa = x + dwconv3x3(x, replicate pad) + b -----------
  // Lane keeps 64 channels of its pixel: run r holds c = r*32 + h*16 + (0..15)
  // -> exactly the B-fragment for QKV k-step r (B map: k = 16*h + j).
  v16h xa[4];
  {
    const float* xb = x + (size_t)(b * 128) * 65536;
#pragma unroll
    for (int r = 0; r < 4; ++r) {
#pragma unroll
      for (int j = 0; j < 16; ++j) {
        const int c = r * 32 + h * 16 + j;
        const float* xc = xb + (size_t)c * 65536;
        float a = xc[gy * 256 + gx] + cpe_b[c];
#pragma unroll
        for (int ky = 0; ky < 3; ++ky) {
          const int yy = clampi(gy + ky - 1, 0, 255);
#pragma unroll
          for (int kx = 0; kx < 3; ++kx) {
            const int xx = clampi(gx + kx - 1, 0, 255);
            a += cpe_w[c * 9 + ky * 3 + kx] * xc[yy * 256 + xx];
          }
        }
        xa[r][j] = (_Float16)a;
      }
    }
  }

  const v8f vzero = {0.f, 0.f, 0.f, 0.f, 0.f, 0.f, 0.f, 0.f};
  v8f o4[8];                       // final output accum: d-tiles x (8 rows)
#pragma unroll
  for (int i = 0; i < 8; ++i) o4[i] = vzero;

  // S-tile pruning: qy == w, so only ky in [w-2,w+2]; 6 even-aligned tiles.
  const int mt0 = clampi(w - 2, 0, 10) & ~1;
  const int nvy = clampi(w + 2, 0, 15) - clampi(w - 2, 0, 15) + 1;
  const int nvx = clampi(n + 2, 0, 15) - clampi(n - 2, 0, 15) + 1;
  const float npad = (float)(25 - nvy * nvx);
  const float scale = 0.17677669529663687f;   // 1/sqrt(32)

  for (int br = 0; br < 2; ++br) {
    // weights for this branch are in flight -> drain + make visible to all
    wait_weight_copy();
    __syncthreads();

    // ---------------- QKV projections (WMMA, K=128) -------------------------
    v8h qreg[4][2];                // Q in registers (f16), [head][c-tile]
#pragma unroll
    for (int hh = 0; hh < 4; ++hh) {
#pragma unroll
      for (int proj = 0; proj < 3; ++proj) {
        v8f acc[2] = {vzero, vzero};
        const _Float16* Wp = W_lds + ((proj * 4 + hh) << 12);
#pragma unroll
        for (int kk = 0; kk < 4; ++kk) {
#pragma unroll
          for (int mt = 0; mt < 2; ++mt) {
            const _Float16* ap = Wp + (mt * 16 + n) * 128 + kk * 32 + h * 16;
            acc[mt] = wmma16(cat16(ap, ap + 8), xa[kk], acc[mt]);
          }
        }
        if (proj == 0) {                     // Q -> registers
#pragma unroll
          for (int mt = 0; mt < 2; ++mt) {
            v8h q;
#pragma unroll
            for (int g = 0; g < 8; ++g) q[g] = (_Float16)acc[mt][g];
            qreg[hh][mt] = q;
          }
        } else if (proj == 1) {              // K -> LDS pixel-major
#pragma unroll
          for (int mt = 0; mt < 2; ++mt) {
            v8h st;
#pragma unroll
            for (int g = 0; g < 8; ++g) st[g] = (_Float16)acc[mt][g];
            *(v8h*)(K_lds + p * KSTR + hh * 32 + mt * 16 + h * 8) = st;
          }
        } else {                             // V -> LDS channel-major
#pragma unroll
          for (int mt = 0; mt < 2; ++mt)
#pragma unroll
            for (int g = 0; g < 8; ++g)
              V_lds[(hh * 32 + mt * 16 + h * 8 + g) * VSTR + p] =
                  (_Float16)acc[mt][g];
        }
      }
    }
    __syncthreads();

    // All waves are done reading W_lds: prefetch branch-1 weights now so the
    // async copy overlaps the whole attention phase.
    if (br == 0) issue_weight_copy(wall, W_lds, 1, tid);

    // ---------------- neighborhood attention + output proj ------------------
#pragma unroll
    for (int hh = 0; hh < 4; ++hh) {
      // B-fragment of Q for this head (lane pair exchange, k = 16*h + j)
      v8h q0 = qreg[hh][0], q1 = qreg[hh][1];
      v8h qx0 = shflx16_h8(q0), qx1 = shflx16_h8(q1);
      v16h qb;
#pragma unroll
      for (int j = 0; j < 8; ++j) {
        qb[j] = h ? qx1[j] : q0[j];
        qb[8 + j] = h ? q1[j] : qx0[j];
      }

      // S = K^T Q over 6 candidate row-tiles
      v8f S[6];
#pragma unroll
      for (int t = 0; t < 6; ++t) {
        const int pk = (mt0 + t) * 16 + n;   // A-matrix row (lane = m)
        const _Float16* kp = K_lds + pk * KSTR + hh * 32 + h * 16;
        S[t] = wmma16(cat16(kp, kp + 8), qb, vzero);
      }

      // mask + softmax (includes zero-score padded neighbors: npad term)
      float mx = -3.0e38f;
#pragma unroll
      for (int t = 0; t < 6; ++t) {
        const int dy = mt0 + t - w;
#pragma unroll
        for (int g = 0; g < 8; ++g) {
          const int dx = h * 8 + g - n;
          const bool valid = (unsigned)(dy + 2) <= 4u && (unsigned)(dx + 2) <= 4u;
          const float s = valid ? S[t][g] * scale : -3.0e38f;
          S[t][g] = s;
          mx = fmaxf(mx, s);
        }
      }
      mx = fmaxf(mx, __shfl_xor(mx, 16));
      mx = (npad > 0.f) ? fmaxf(mx, 0.f) : mx;
      float l = 0.f;
#pragma unroll
      for (int t = 0; t < 6; ++t)
#pragma unroll
        for (int g = 0; g < 8; ++g) {
          const float e = (S[t][g] > -1.0e37f) ? __expf(S[t][g] - mx) : 0.f;
          S[t][g] = e;
          l += e;
        }
      l += __shfl_xor(l, 16);
      l += npad * __expf(-mx);
      const float inv = 1.0f / l;
#pragma unroll
      for (int t = 0; t < 6; ++t)
#pragma unroll
        for (int g = 0; g < 8; ++g) S[t][g] *= inv;

      // O = V * P   (P fed straight from S accumulators via lane exchange)
      v8f oa[2] = {vzero, vzero};
#pragma unroll
      for (int t3 = 0; t3 < 3; ++t3) {
        v8f sA = S[2 * t3], sB = S[2 * t3 + 1];
        float xA[8], xB[8];
#pragma unroll
        for (int g = 0; g < 8; ++g) {
          xA[g] = __shfl_xor(sA[g], 16);
          xB[g] = __shfl_xor(sB[g], 16);
        }
        v16h pf;
#pragma unroll
        for (int j = 0; j < 8; ++j) {
          pf[j] = (_Float16)(h ? xB[j] : sA[j]);
          pf[8 + j] = (_Float16)(h ? sB[j] : xA[j]);
        }
        const int pkb = (mt0 + 2 * t3) * 16;
#pragma unroll
        for (int mo = 0; mo < 2; ++mo) {
          const _Float16* vp = V_lds + (hh * 32 + mo * 16 + n) * VSTR + pkb + h * 16;
          oa[mo] = wmma16(cat16(vp, vp + 8), pf, oa[mo]);
        }
      }

      // Wo GEMM: o4[d][:] += Wo[br][hh] (128x32) x O_hh (32x16), O from regs.
      // Wo A-fragments come from global (L0-resident); W_lds may be in flight.
      float oxA[8], oxB[8];
#pragma unroll
      for (int g = 0; g < 8; ++g) {
        oxA[g] = __shfl_xor(oa[0][g], 16);
        oxB[g] = __shfl_xor(oa[1][g], 16);
      }
      v16h of;
#pragma unroll
      for (int j = 0; j < 8; ++j) {
        of[j] = (_Float16)(h ? oxB[j] : oa[0][j]);
        of[8 + j] = (_Float16)(h ? oa[1][j] : oxA[j]);
      }
      const _Float16* Wop = wall + 98304 + ((br * 4 + hh) << 12);
#pragma unroll
      for (int md = 0; md < 8; ++md) {
        const _Float16* ap = Wop + (md * 16 + n) * 32 + h * 16;
        o4[md] = wmma16(cat16(ap, ap + 8), of, o4[md]);
      }
    }
    __syncthreads();
  }

  // ---------------- write output: total / BRANCHES --------------------------
  float* ob = out + (size_t)(b * 128) * 65536 + gy * 256 + gx;
#pragma unroll
  for (int md = 0; md < 8; ++md)
#pragma unroll
    for (int g = 0; g < 8; ++g) {
      const int d = md * 16 + h * 8 + g;
      ob[(size_t)d * 65536] = o4[md][g] * 0.5f;
    }
}

// ---------------------------------------------------------------------------
extern "C" void kernel_launch(void* const* d_in, const int* in_sizes, int n_in,
                              void* d_out, int out_size, void* d_ws,
                              size_t ws_size, hipStream_t stream) {
  const float* x = (const float*)d_in[0];
  const float* cpe_w = (const float*)d_in[1];
  const float* cpe_b = (const float*)d_in[2];
  const float* Wq = (const float*)d_in[3];
  const float* Wk = (const float*)d_in[4];
  const float* Wv = (const float*)d_in[5];
  const float* Wo = (const float*)d_in[6];
  float* outp = (float*)d_out;
  _Float16* wf = (_Float16*)d_ws;   // needs 131072 halves = 256 KiB

  (void)in_sizes; (void)n_in; (void)out_size; (void)ws_size;
  (void)hipFuncSetAttribute((const void*)fwa_main,
                            hipFuncAttributeMaxDynamicSharedMemorySize,
                            (int)SMEM_BYTES);

  fwa_cvt_weights<<<512, 256, 0, stream>>>(Wq, Wk, Wv, Wo, wf);
  fwa_main<<<1024, 512, SMEM_BYTES, stream>>>(x, cpe_w, cpe_b, wf, outp);
}
